// IzhikevichNet_90872918049236
// MI455X (gfx1250) — matrix operand order repaired and checked by reading, not compiled, not run on hardware
//
#include <hip/hip_runtime.h>
#include <hip/hip_bf16.h>

typedef __attribute__((ext_vector_type(16))) __bf16 v16bf;
typedef __attribute__((ext_vector_type(8)))  __bf16 v8bf;
typedef __attribute__((ext_vector_type(8)))  float  v8f;

#define NUM_INPUT  784
#define KPAD       800
#define NUM_HIDDEN 100
#define NPAD       112
#define NUM_OUTPUT 10
#define NUM_STEPS  100
#define BATCH      2048
#define ROWS       (BATCH * NUM_STEPS)   // 204800 flattened (b,t) rows

// ---------------------------------------------------------------------------
// Kernel 0: split W1 (fp32 [100][784]) into bf16 hi/lo, zero-padded [112][800]
// ---------------------------------------------------------------------------
__global__ __launch_bounds__(256) void prep_w1(const float* __restrict__ W1,
                                               __bf16* __restrict__ Whi,
                                               __bf16* __restrict__ Wlo) {
  int idx = blockIdx.x * 256 + threadIdx.x;
  if (idx >= NPAD * KPAD) return;
  int n = idx / KPAD;
  int k = idx - n * KPAD;
  float w = (n < NUM_HIDDEN && k < NUM_INPUT) ? W1[n * NUM_INPUT + k] : 0.0f;
  __bf16 hi = (__bf16)w;
  Whi[idx] = hi;
  Wlo[idx] = (__bf16)(w - (float)hi);
}

// ---------------------------------------------------------------------------
// A-fragment loader: fp32 row -> split-bf16 hi/lo WMMA fragment
// 16-bit A 16x32 layout: lane half g: halves[0..7] = K=k0+8g.. ;
//                                     halves[8..15] = K=k0+8g+16..
// ---------------------------------------------------------------------------
__device__ __forceinline__ void load_a_frag(const float* __restrict__ xrow,
                                            int kbA, v16bf& ahi, v16bf& alo) {
  float f[16];
  {
    float4 a0 = *(const float4*)(xrow + kbA);
    float4 a1 = *(const float4*)(xrow + kbA + 4);
    f[0]=a0.x; f[1]=a0.y; f[2]=a0.z; f[3]=a0.w;
    f[4]=a1.x; f[5]=a1.y; f[6]=a1.z; f[7]=a1.w;
  }
  if (kbA + 24 <= NUM_INPUT) {
    float4 a2 = *(const float4*)(xrow + kbA + 16);
    float4 a3 = *(const float4*)(xrow + kbA + 20);
    f[8]=a2.x;  f[9]=a2.y;  f[10]=a2.z; f[11]=a2.w;
    f[12]=a3.x; f[13]=a3.y; f[14]=a3.z; f[15]=a3.w;
  } else {
#pragma unroll
    for (int i = 8; i < 16; ++i) f[i] = 0.0f;
  }
#pragma unroll
  for (int i = 0; i < 16; ++i) {
    __bf16 h = (__bf16)f[i];
    ahi[i] = h;
    alo[i] = (__bf16)(f[i] - (float)h);
  }
}

// ---------------------------------------------------------------------------
// Kernel 1: CUR1[r][h] = X[r][:] dot W1[h][:]   (bias added later)
// Split-bf16 GEMM: acc += Ahi*Bhi + Ahi*Blo + Alo*Bhi (fp32 accumulate)
// One wave per 32-row strip (2 M-tiles, B fragments reused); 7 N-tiles.
// ---------------------------------------------------------------------------
__global__ __launch_bounds__(256) void gemm_cur1(const float*  __restrict__ X,
                                                 const __bf16* __restrict__ Whi,
                                                 const __bf16* __restrict__ Wlo,
                                                 float* __restrict__ CUR1) {
  const int lane  = threadIdx.x & 31;
  const int wave  = threadIdx.x >> 5;
  const int strip = blockIdx.x * 8 + wave;       // 6400 strips of 32 rows
  const int m0    = strip * 32;
  const int half  = lane >> 4;                   // 0 or 1 (half-wave)
  const int l16   = lane & 15;

  v8f acc[2][7];
#pragma unroll
  for (int s = 0; s < 2; ++s)
#pragma unroll
    for (int i = 0; i < 7; ++i) acc[s][i] = (v8f){0,0,0,0,0,0,0,0};

  const float* xrow0 = X + (size_t)(m0 + l16) * NUM_INPUT;
  const float* xrow1 = X + (size_t)(m0 + 16 + l16) * NUM_INPUT;

  for (int k0 = 0; k0 < KPAD; k0 += 32) {
    const int kbA = k0 + half * 8;
    v16bf ahi0, alo0, ahi1, alo1;
    load_a_frag(xrow0, kbA, ahi0, alo0);
    load_a_frag(xrow1, kbA, ahi1, alo1);

    // 16-bit B 32x16 layout: lane half g: [0..7] = K=k0+8g.., [8..15] = K=k0+8g+16..
    const int kbB = k0 + half * 8;
#pragma unroll
    for (int nt = 0; nt < 7; ++nt) {
      const size_t boff = (size_t)(nt * 16 + l16) * KPAD + kbB;
      v8bf bh0 = *(const v8bf*)(Whi + boff);
      v8bf bh1 = *(const v8bf*)(Whi + boff + 16);
      v8bf bl0 = *(const v8bf*)(Wlo + boff);
      v8bf bl1 = *(const v8bf*)(Wlo + boff + 16);
      v16bf bhi = __builtin_shufflevector(bh0, bh1, 0,1,2,3,4,5,6,7,8,9,10,11,12,13,14,15);
      v16bf blo = __builtin_shufflevector(bl0, bl1, 0,1,2,3,4,5,6,7,8,9,10,11,12,13,14,15);
      acc[0][nt] = __builtin_amdgcn_wmma_f32_16x16x32_bf16(false, ahi0, false, bhi,
                                                           (short)0, acc[0][nt], false, false);
      acc[0][nt] = __builtin_amdgcn_wmma_f32_16x16x32_bf16(false, ahi0, false, blo,
                                                           (short)0, acc[0][nt], false, false);
      acc[0][nt] = __builtin_amdgcn_wmma_f32_16x16x32_bf16(false, alo0, false, bhi,
                                                           (short)0, acc[0][nt], false, false);
      acc[1][nt] = __builtin_amdgcn_wmma_f32_16x16x32_bf16(false, ahi1, false, bhi,
                                                           (short)0, acc[1][nt], false, false);
      acc[1][nt] = __builtin_amdgcn_wmma_f32_16x16x32_bf16(false, ahi1, false, blo,
                                                           (short)0, acc[1][nt], false, false);
      acc[1][nt] = __builtin_amdgcn_wmma_f32_16x16x32_bf16(false, alo1, false, bhi,
                                                           (short)0, acc[1][nt], false, false);
    }
  }

  // ---- store C: VGPR r -> (M = mbase + r, N = nt*16 + l16) ----
#pragma unroll
  for (int s = 0; s < 2; ++s) {
    const int mbase = m0 + s * 16 + half * 8;
#pragma unroll
    for (int nt = 0; nt < 7; ++nt) {
      int n = nt * 16 + l16;
      if (n < NUM_HIDDEN) {
#pragma unroll
        for (int r = 0; r < 8; ++r)
          CUR1[(size_t)(mbase + r) * NUM_HIDDEN + n] = acc[s][nt][r];
      }
    }
  }
}

// ---------------------------------------------------------------------------
// Kernel 2: sequential Izhikevich dynamics. One wave per batch sample.
// Lane owns hidden neurons {lane, lane+32, lane+64, lane+96} (h<100).
// ---------------------------------------------------------------------------
__global__ __launch_bounds__(256) void dynamics(const float* __restrict__ CUR1,
                                                const float* __restrict__ b1,
                                                const float* __restrict__ W2,
                                                const float* __restrict__ b2,
                                                float* __restrict__ out) {
  const int lane = threadIdx.x & 31;
  const int wave = threadIdx.x >> 5;
  const int bsmp = blockIdx.x * 8 + wave;        // 0..2047

  const float A = 0.02f, BB = 0.2f, C = -65.0f, D = 8.0f, THR = 0.03f;

  // preload per-lane constants
  float b1v[4];
  float w2v[10][4];
#pragma unroll
  for (int j = 0; j < 4; ++j) {
    int h = lane + 32 * j;
    bool ok = (h < NUM_HIDDEN);
    b1v[j] = ok ? b1[h] : 0.0f;
#pragma unroll
    for (int o = 0; o < 10; ++o)
      w2v[o][j] = ok ? W2[o * NUM_HIDDEN + h] : 0.0f;
  }
  float b2v = (lane < 10) ? b2[lane] : 0.0f;

  float v1[4], u1[4];
#pragma unroll
  for (int j = 0; j < 4; ++j) { v1[j] = -70.0f; u1[j] = -15.0f; }
  float v2 = -70.0f, u2 = -15.0f;

  for (int t = 0; t < NUM_STEPS; ++t) {
    const size_t r = (size_t)bsmp * NUM_STEPS + t;
    float p[10];
#pragma unroll
    for (int o = 0; o < 10; ++o) p[o] = 0.0f;

#pragma unroll
    for (int j = 0; j < 4; ++j) {
      int h = lane + 32 * j;
      float I = (h < NUM_HIDDEN) ? (CUR1[r * NUM_HIDDEN + h] + b1v[j]) : 0.0f;
      float vn = v1[j] + 0.04f * v1[j] * v1[j] + 5.0f * v1[j] + 140.0f - u1[j] + I;
      float un = u1[j] + A * (BB * v1[j] - u1[j]);
      float spk = (vn >= THR) ? 1.0f : 0.0f;
      v1[j] = (spk > 0.0f) ? C : vn;
      u1[j] = (spk > 0.0f) ? (un + D) : un;
#pragma unroll
      for (int o = 0; o < 10; ++o) p[o] = fmaf(spk, w2v[o][j], p[o]);
    }

    // wave32 reduction: all lanes end with the full sums
#pragma unroll
    for (int o = 0; o < 10; ++o) {
      float s = p[o];
      s += __shfl_xor(s, 16, 32);
      s += __shfl_xor(s, 8, 32);
      s += __shfl_xor(s, 4, 32);
      s += __shfl_xor(s, 2, 32);
      s += __shfl_xor(s, 1, 32);
      p[o] = s;
    }

    // lane o (< 10) picks its own current
    float s = p[0];
#pragma unroll
    for (int o = 1; o < 10; ++o)
      if (lane == o) s = p[o];

    if (lane < 10) {
      float I2 = s + b2v;
      float vn = v2 + 0.04f * v2 * v2 + 5.0f * v2 + 140.0f - u2 + I2;
      float un = u2 + A * (BB * v2 - u2);
      float spk = (vn >= THR) ? 1.0f : 0.0f;
      v2 = (spk > 0.0f) ? C : vn;
      u2 = (spk > 0.0f) ? (un + D) : un;
      size_t ob = (size_t)t * (BATCH * NUM_OUTPUT) + (size_t)bsmp * NUM_OUTPUT + lane;
      out[ob] = spk;                                        // spk2_rec
      out[(size_t)NUM_STEPS * BATCH * NUM_OUTPUT + ob] = v2; // mem2_rec
    }
  }
}

// ---------------------------------------------------------------------------
extern "C" void kernel_launch(void* const* d_in, const int* in_sizes, int n_in,
                              void* d_out, int out_size, void* d_ws, size_t ws_size,
                              hipStream_t stream) {
  const float* x  = (const float*)d_in[0];
  const float* W1 = (const float*)d_in[1];
  const float* b1 = (const float*)d_in[2];
  const float* W2 = (const float*)d_in[3];
  const float* b2 = (const float*)d_in[4];
  float* out = (float*)d_out;

  char* ws = (char*)d_ws;
  float* CUR1 = (float*)ws;                                 // 204800*100 f32 = 81.92 MB
  size_t cur1_bytes = (size_t)ROWS * NUM_HIDDEN * sizeof(float);
  __bf16* Whi = (__bf16*)(ws + cur1_bytes);                 // [112][800] bf16
  __bf16* Wlo = Whi + (size_t)NPAD * KPAD;

  prep_w1<<<(NPAD * KPAD + 255) / 256, 256, 0, stream>>>(W1, Whi, Wlo);
  gemm_cur1<<<(ROWS / 32) / 8, 256, 0, stream>>>(x, Whi, Wlo, CUR1);
  dynamics<<<BATCH / 8, 256, 0, stream>>>(CUR1, b1, W2, b2, out);
}
